// NoisyGRUSeq2SeqWithFeatures_21157008900487
// MI455X (gfx1250) — matrix operand order repaired and checked by reading, not compiled, not run on hardware
//
#include <hip/hip_runtime.h>

// ---------------------------------------------------------------------------
// Types for CDNA5 WMMA (wave32): A/B = 16 x bf16 (8 VGPRs), C/D = 8 x f32.
// ---------------------------------------------------------------------------
typedef __attribute__((ext_vector_type(16))) __bf16 bf16x16;
typedef __attribute__((ext_vector_type(8)))  float  f32x8;

union ABFrag { bf16x16 v; uint4 q[2]; };

__device__ __forceinline__ unsigned short f2bf(float f) {
  // round-to-nearest-even fp32 -> bf16 bits
  unsigned u = __float_as_uint(f);
  u += 0x7FFFu + ((u >> 16) & 1u);
  return (unsigned short)(u >> 16);
}

// Load one K-step (32) of fragments: A row chunk + 4 B column chunks.
// A (16-bit 16x32 ISA layout): lanes 0-15 K{0..7,16..23}, lanes 16-31
// K{8..15,24..31} -> two b128 at +0/+32B from koffA.
// B (from [N][K] transposed weights): same K order as A -> two b128 at
// +0/+32B from koffB.
__device__ __forceinline__ void load_frags(const unsigned short* __restrict__ arow,
                                           const unsigned short* const* brow,
                                           int k, ABFrag& a, ABFrag* b) {
  a.q[0] = *(const uint4*)(arow + k);
  a.q[1] = *(const uint4*)(arow + k + 16);
#pragma unroll
  for (int s = 0; s < 4; ++s) {
    b[s].q[0] = *(const uint4*)(brow[s] + k);
    b[s].q[1] = *(const uint4*)(brow[s] + k + 16);
  }
}

// ---------------------------------------------------------------------------
// Fused WMMA GEMM: D[m,n] = A[256,K](bf16) @ Wt[N,K]^T(bf16) + bias, with a
// mode-selected GRU epilogue. Each wave computes a 16x64 tile (4 WMMAs /
// K-step of 32, A fragment reused), software-pipelined so loads for step
// k+1 overlap WMMAs for step k.
//   mode 0: gates   v=sigmoid(.);  n<outs: bf16(r*h) -> bf1(+off1);
//                                  n>=outs: z (f32)
//   mode 1: cand    c=tanh(.); hn = z*h+(1-z)*c masked by (t<len);
//                   h<-hn (f32); bf16(hn) -> bf1(+off1) [, bf2, bf3]
//   mode 2: dense   fout = tanh(.)
// ---------------------------------------------------------------------------
__global__ __launch_bounds__(256) void wmma_gemm_gru(
    const unsigned short* __restrict__ A, int lda,
    const unsigned short* __restrict__ Wt, int K, int N,
    const float* __restrict__ bias,
    int mode, int t,
    float* __restrict__ h, int outs,
    float* __restrict__ z,
    const int* __restrict__ lens,
    unsigned short* __restrict__ bf1, int stride1, int off1,
    unsigned short* __restrict__ bf2,
    unsigned short* __restrict__ bf3, int stride23,
    float* __restrict__ fout)
{
  const int lane   = threadIdx.x & 31;
  const int wave   = threadIdx.x >> 5;
  const int tilesN = N >> 6;
  const int gw     = blockIdx.x * (blockDim.x >> 5) + wave;
  const int tm     = gw / tilesN;
  const int tn     = gw % tilesN;
  if ((tm << 4) >= 256) return;          // wave-uniform; EXEC stays all-1s

  const int mrow  = (tm << 4) + (lane & 15);
  const int koffA = (lane >> 4) << 3;     // 0 or 8
  const int koffB = (lane >> 4) << 3;     // 0 or 8
  const int ncol  = (tn << 6) + (lane & 15);

  const unsigned short* arow = A + (size_t)mrow * lda + koffA;
  const unsigned short* brow[4];
#pragma unroll
  for (int s = 0; s < 4; ++s)
    brow[s] = Wt + (size_t)(ncol + 16 * s) * K + koffB;

  f32x8 acc[4] = {{}, {}, {}, {}};

  // -------- software-pipelined K loop (register double buffer) ------------
  ABFrag a_cur, b_cur[4];
  load_frags(arow, brow, 0, a_cur, b_cur);

  for (int k = 32; k < K; k += 32) {
    ABFrag a_nxt, b_nxt[4];
    load_frags(arow, brow, k, a_nxt, b_nxt);       // in flight during WMMAs
#pragma unroll
    for (int s = 0; s < 4; ++s)
      acc[s] = __builtin_amdgcn_wmma_f32_16x16x32_bf16(
                   false, a_cur.v, false, b_cur[s].v, (short)0, acc[s], false, false);
    a_cur = a_nxt;
#pragma unroll
    for (int s = 0; s < 4; ++s) b_cur[s] = b_nxt[s];
  }
#pragma unroll
  for (int s = 0; s < 4; ++s)
    acc[s] = __builtin_amdgcn_wmma_f32_16x16x32_bf16(
                 false, a_cur.v, false, b_cur[s].v, (short)0, acc[s], false, false);

  // C/D f32 16x16 layout: VGPR e, lanes 0-15 -> M=e, N=lane;
  // lanes 16-31 -> M=e+8, N=lane-16.
  const int mbase = (tm << 4) + ((lane >> 4) << 3);
#pragma unroll
  for (int s = 0; s < 4; ++s) {
    const int n  = (tn << 6) + (s << 4) + (lane & 15);
    const float bv = bias[n];
#pragma unroll
    for (int e = 0; e < 8; ++e) {
      const int m = mbase + e;
      float v = acc[s][e] + bv;
      if (mode == 0) {
        v = 1.0f / (1.0f + __expf(-v));
        if (n < outs) {
          float rh = v * h[(size_t)m * outs + n];
          bf1[(size_t)m * stride1 + off1 + n] = f2bf(rh);
        } else {
          z[(size_t)m * outs + (n - outs)] = v;
        }
      } else if (mode == 1) {
        float c  = tanhf(v);
        float zz = z[(size_t)m * outs + n];
        float ho = h[(size_t)m * outs + n];
        float hn = zz * ho + (1.0f - zz) * c;
        hn = (t < lens[m]) ? hn : ho;
        h[(size_t)m * outs + n] = hn;
        unsigned short hb = f2bf(hn);
        bf1[(size_t)m * stride1 + off1 + n] = hb;
        if (bf2) {
          bf2[(size_t)m * stride23 + n] = hb;
          bf3[(size_t)m * stride23 + n] = hb;
        }
      } else {
        fout[(size_t)m * N + n] = tanhf(v);
      }
    }
  }
}

// fp32 [K][N] -> bf16 [N][K] (transpose so B fragments are contiguous in K)
__global__ void conv_tr(const float* __restrict__ W, unsigned short* __restrict__ Wt,
                        int K, int N) {
  size_t i = (size_t)blockIdx.x * blockDim.x + threadIdx.x;
  if (i >= (size_t)K * N) return;
  int n = (int)(i / K);
  int k = (int)(i % K);
  Wt[i] = f2bf(W[(size_t)k * N + n]);
}

__global__ void zero_f32(float* __restrict__ p, int n) {
  int i = blockIdx.x * blockDim.x + threadIdx.x;
  if (i < n) p[i] = 0.0f;
}
__global__ void zero_u16(unsigned short* __restrict__ p, int n) {
  int i = blockIdx.x * blockDim.x + threadIdx.x;
  if (i < n) p[i] = 0;
}

// per-timestep embedding gather -> bf16 x-part of layer-0 A buffers
__global__ void embed_kernel(const int* __restrict__ seqs, const float* __restrict__ emb,
                             unsigned short* __restrict__ xh0, unsigned short* __restrict__ xrh0,
                             int t, int T) {
  int b = blockIdx.x;
  int c = threadIdx.x;                 // 32 = CHAR_EMB
  int tok = seqs[b * T + t];
  unsigned short v = f2bf(emb[tok * 32 + c]);
  xh0 [b * 544 + c] = v;
  xrh0[b * 544 + c] = v;
}

__global__ void build_hcat(const float* __restrict__ h0, const float* __restrict__ h1,
                           const float* __restrict__ h2, unsigned short* __restrict__ hcat) {
  int i = blockIdx.x * blockDim.x + threadIdx.x;
  if (i >= 256 * 3584) return;
  int b = i / 3584, c = i % 3584;
  float v = (c < 512)  ? h0[b * 512 + c]
          : (c < 1536) ? h1[b * 1024 + (c - 512)]
                       : h2[b * 2048 + (c - 1536)];
  hcat[i] = f2bf(v);
}

extern "C" void kernel_launch(void* const* d_in, const int* in_sizes, int n_in,
                              void* d_out, int out_size, void* d_ws, size_t ws_size,
                              hipStream_t stream) {
  const int*   seqs = (const int*)  d_in[0];
  const int*   lens = (const int*)  d_in[1];
  const float* emb  = (const float*)d_in[2];
  const float* gk[3] = {(const float*)d_in[3],  (const float*)d_in[7],  (const float*)d_in[11]};
  const float* gb[3] = {(const float*)d_in[4],  (const float*)d_in[8],  (const float*)d_in[12]};
  const float* ck[3] = {(const float*)d_in[5],  (const float*)d_in[9],  (const float*)d_in[13]};
  const float* cb[3] = {(const float*)d_in[6],  (const float*)d_in[10], (const float*)d_in[14]};
  const float* dw = (const float*)d_in[15];
  const float* db = (const float*)d_in[16];
  float* out = (float*)d_out;

  const int B = 256, T = 128;
  const int ins[3]  = {32, 512, 1024};
  const int outs[3] = {512, 1024, 2048};

  // ---- carve workspace (~67 MB) -------------------------------------------
  char* p = (char*)d_ws;
  auto carve = [&](size_t bytes) -> void* {
    void* r = (void*)p;
    p += (bytes + 255) & ~(size_t)255;
    return r;
  };
  unsigned short *gkt[3], *ckt[3];
  for (int l = 0; l < 3; ++l) {
    size_t kk = (size_t)ins[l] + outs[l];
    gkt[l] = (unsigned short*)carve(kk * 2 * outs[l] * 2);
    ckt[l] = (unsigned short*)carve(kk * outs[l] * 2);
  }
  unsigned short* dwt = (unsigned short*)carve((size_t)3584 * 512 * 2);
  unsigned short *xh[3], *xrh[3];
  for (int l = 0; l < 3; ++l) {
    size_t kk = (size_t)ins[l] + outs[l];
    xh[l]  = (unsigned short*)carve((size_t)B * kk * 2);
    xrh[l] = (unsigned short*)carve((size_t)B * kk * 2);
  }
  unsigned short* hcat = (unsigned short*)carve((size_t)B * 3584 * 2);
  float *h[3], *z[3];
  for (int l = 0; l < 3; ++l) {
    h[l] = (float*)carve((size_t)B * outs[l] * 4);
    z[l] = (float*)carve((size_t)B * outs[l] * 4);
  }

  // ---- one-time (per launch) weight conversion: fp32 [K][N] -> bf16 [N][K]
  for (int l = 0; l < 3; ++l) {
    int kk = ins[l] + outs[l];
    size_t ng = (size_t)kk * 2 * outs[l];
    conv_tr<<<(unsigned)((ng + 255) / 256), 256, 0, stream>>>(gk[l], gkt[l], kk, 2 * outs[l]);
    size_t nc = (size_t)kk * outs[l];
    conv_tr<<<(unsigned)((nc + 255) / 256), 256, 0, stream>>>(ck[l], ckt[l], kk, outs[l]);
  }
  {
    size_t nd = (size_t)3584 * 512;
    conv_tr<<<(unsigned)((nd + 255) / 256), 256, 0, stream>>>(dw, dwt, 3584, 512);
  }

  // ---- zero recurrent state + bf16 A buffers ------------------------------
  for (int l = 0; l < 3; ++l) {
    int nh = B * outs[l];
    zero_f32<<<(nh + 255) / 256, 256, 0, stream>>>(h[l], nh);
    int nx = B * (ins[l] + outs[l]);
    zero_u16<<<(nx + 255) / 256, 256, 0, stream>>>(xh[l],  nx);
    zero_u16<<<(nx + 255) / 256, 256, 0, stream>>>(xrh[l], nx);
  }

  auto blocks = [](int N) { return (16 * (N >> 6)) >> 3; };  // waves/8, 8 waves per 256-thr block

  // ---- time loop ----------------------------------------------------------
  for (int t = 0; t < T; ++t) {
    embed_kernel<<<B, 32, 0, stream>>>(seqs, emb, xh[0], xrh[0], t, T);
    for (int l = 0; l < 3; ++l) {
      int kk = ins[l] + outs[l];
      int Ng = 2 * outs[l];
      // gates: v = sigmoid([x,h]@Kg+bg); epilogue emits bf16(r*h) and z
      wmma_gemm_gru<<<blocks(Ng), 256, 0, stream>>>(
          xh[l], kk, gkt[l], kk, Ng, gb[l], /*mode*/0, t,
          h[l], outs[l], z[l], lens,
          xrh[l], kk, ins[l], nullptr, nullptr, 0, nullptr);
      // candidate + masked state update; scatter bf16(h_new) forward
      int Nc = outs[l];
      unsigned short* nb2 = (l < 2) ? xh[l + 1]  : nullptr;
      unsigned short* nb3 = (l < 2) ? xrh[l + 1] : nullptr;
      int st23 = (l < 2) ? (ins[l + 1] + outs[l + 1]) : 0;
      wmma_gemm_gru<<<blocks(Nc), 256, 0, stream>>>(
          xrh[l], kk, ckt[l], kk, Nc, cb[l], /*mode*/1, t,
          h[l], outs[l], z[l], lens,
          xh[l], kk, ins[l], nb2, nb3, st23, nullptr);
    }
  }

  // ---- final dense: tanh(concat(h) @ dw + db) -----------------------------
  {
    int n = B * 3584;
    build_hcat<<<(n + 255) / 256, 256, 0, stream>>>(h[0], h[1], h[2], hcat);
  }
  wmma_gemm_gru<<<blocks(512), 256, 0, stream>>>(
      hcat, 3584, dwt, 3584, 512, db, /*mode*/2, 0,
      nullptr, 0, nullptr, nullptr, nullptr, 0, 0, nullptr, nullptr, 0, out);

  (void)in_sizes; (void)n_in; (void)out_size; (void)ws_size;
}
